// GraphAttentionLayer_14310831030374
// MI455X (gfx1250) — matrix operand order repaired and checked by reading, not compiled, not run on hardware
//
#include <hip/hip_runtime.h>

typedef __attribute__((ext_vector_type(16))) __bf16 v16bf;
typedef __attribute__((ext_vector_type(8)))  __bf16 v8bf;
typedef __attribute__((ext_vector_type(8)))  float  v8f;

#define NN    8192
#define INF_  256
#define OUTF  128
#define ALPHA 0.2f

#define SCHED_FENCE() __builtin_amdgcn_sched_barrier(0)

// workspace byte offsets
#define OFF_WT    0u          // WT  : [128][256] bf16  = 65536 B
#define OFF_HBT   65536u      // hbT : [128][8192] bf16 = 2097152 B
#define OFF_FSRC  2162688u    // f_src : 8192 f32
#define OFF_FDST  2195456u    // f_dst : 8192 f32
#define OFF_GMAX  2228224u    // 1 f32

// ---------------------------------------------------------------------------
// K1: transpose W [256x128] f32 -> WT [128][256] bf16 (B-fragment friendly)
// ---------------------------------------------------------------------------
__global__ void gat_prep_wt(const float* __restrict__ W, __bf16* __restrict__ WT) {
    int idx = blockIdx.x * blockDim.x + threadIdx.x;   // coalesced read of W
    int k = idx >> 7;          // 0..255
    int n = idx & 127;         // 0..127
    WT[n * INF_ + k] = (__bf16)W[idx];
}

// ---------------------------------------------------------------------------
// K2: h = X@W via WMMA bf16; emit hbT (transposed bf16 h) + f_src/f_dst
// ---------------------------------------------------------------------------
__global__ void __launch_bounds__(128)
gat_gemm_h(const float* __restrict__ X, const __bf16* __restrict__ WT,
           const float* __restrict__ av,
           __bf16* __restrict__ hbT, float* __restrict__ f_src, float* __restrict__ f_dst) {
    const int lane = threadIdx.x & 31;
    const int wv   = threadIdx.x >> 5;
    const int tile = blockIdx.x * 4 + wv;      // 0..511
    const int m0   = tile << 4;
    const int l16  = lane & 15;
    const bool hi  = lane >= 16;
    const int kofA = hi ? 16 : 0;              // A-fragment K sub-offset
    const int kofB = hi ? 16 : 0;              // B-fragment K sub-offset
    const int rowA = m0 + l16;

    const v8f zero8 = {0.f,0.f,0.f,0.f,0.f,0.f,0.f,0.f};
    v8f C[8];
#pragma unroll
    for (int t = 0; t < 8; ++t) C[t] = zero8;

#pragma unroll 1
    for (int k0 = 0; k0 < INF_; k0 += 32) {
        const float* xp = X + (size_t)rowA * INF_ + k0 + kofA;
        float4 x0 = *(const float4*)(xp);
        float4 x1 = *(const float4*)(xp + 4);
        float4 x2 = *(const float4*)(xp + 8);
        float4 x3 = *(const float4*)(xp + 12);
        // all 8 B tiles batched: distinct registers, progressive waits
        v16bf Bf[8];
#pragma unroll
        for (int t = 0; t < 8; ++t)
            Bf[t] = *(const v16bf*)(WT + (size_t)(t * 16 + l16) * INF_ + k0 + kofB);
        v16bf A;
        A[0]=(__bf16)x0.x;  A[1]=(__bf16)x0.y;  A[2]=(__bf16)x0.z;  A[3]=(__bf16)x0.w;
        A[4]=(__bf16)x1.x;  A[5]=(__bf16)x1.y;  A[6]=(__bf16)x1.z;  A[7]=(__bf16)x1.w;
        A[8]=(__bf16)x2.x;  A[9]=(__bf16)x2.y;  A[10]=(__bf16)x2.z; A[11]=(__bf16)x2.w;
        A[12]=(__bf16)x3.x; A[13]=(__bf16)x3.y; A[14]=(__bf16)x3.z; A[15]=(__bf16)x3.w;
        SCHED_FENCE();
#pragma unroll
        for (int t = 0; t < 8; ++t)
            C[t] = __builtin_amdgcn_wmma_f32_16x16x32_bf16(
                false, A, false, Bf[t], (short)0, C[t], false, false);
        SCHED_FENCE();
    }

    // epilogue: write hbT transposed, fold a-vector dots into f_src/f_dst
    float fs[8], fd[8];
#pragma unroll
    for (int v = 0; v < 8; ++v) { fs[v] = 0.f; fd[v] = 0.f; }

#pragma unroll
    for (int t = 0; t < 8; ++t) {
        const int n = t * 16 + l16;            // D-layout: lane%16 = column
        const float a1 = av[n];
        const float a2 = av[OUTF + n];
        v8bf hb;
#pragma unroll
        for (int v = 0; v < 8; ++v) {
            float c = C[t][v];                 // row = v + (hi?8:0)
            fs[v] += c * a1;
            fd[v] += c * a2;
            hb[v] = (__bf16)c;
        }
        *(v8bf*)(hbT + (size_t)n * NN + m0 + (hi ? 8 : 0)) = hb;  // 16B contiguous
    }

#pragma unroll
    for (int v = 0; v < 8; ++v) {
#pragma unroll
        for (int msk = 1; msk < 16; msk <<= 1) {
            fs[v] += __shfl_xor(fs[v], msk, 32);
            fd[v] += __shfl_xor(fd[v], msk, 32);
        }
    }
    if (l16 == 0) {                            // lanes 0 and 16
        const int base = m0 + (hi ? 8 : 0);
#pragma unroll
        for (int v = 0; v < 8; ++v) { f_src[base + v] = fs[v]; f_dst[base + v] = fd[v]; }
    }
}

// ---------------------------------------------------------------------------
// K3: gmax = max_j f_dst[j]  (single block tree reduction)
// ---------------------------------------------------------------------------
__global__ void __launch_bounds__(1024)
gat_max_fdst(const float* __restrict__ f_dst, float* __restrict__ gmax) {
    __shared__ float red[1024];
    float m = -3.4e38f;
    for (int i = threadIdx.x; i < NN; i += 1024) m = fmaxf(m, f_dst[i]);
    red[threadIdx.x] = m;
    __syncthreads();
    for (int s = 512; s > 0; s >>= 1) {
        if (threadIdx.x < s) red[threadIdx.x] = fmaxf(red[threadIdx.x], red[threadIdx.x + s]);
        __syncthreads();
    }
    if (threadIdx.x == 0) *gmax = red[0];
}

// ---------------------------------------------------------------------------
// K4: fused masked-softmax attention GEMM, single pass over adj.
//     One block (8 waves) per 16-row tile; each wave owns a 1024-column
//     j-chunk (8-way split -> 4096 waves for HBM latency hiding).
//     Scheduling fences pin the pipeline:
//       [B loads (L2) + LDS reads] | [adj loads (HBM)] | [exp VALU] | [WMMA]
//     so waiting on B keeps the adj loads in flight for a full iteration.
// ---------------------------------------------------------------------------
__global__ void __launch_bounds__(256)
gat_attn(const int* __restrict__ adj, const __bf16* __restrict__ hbT,
         const float* __restrict__ f_src, const float* __restrict__ f_dstg,
         const float* __restrict__ gmaxp, float* __restrict__ out) {
    // phase 1: smem[0..8191] = f_dst staging
    // phase 2: smem[0..16383] = 8 partial 16x128 C tiles
    __shared__ float smem[16384];
    __shared__ float psums[128];               // [wave][row]
    {
        const float4* s4 = (const float4*)f_dstg;
        float4* d4 = (float4*)smem;
        for (int i = threadIdx.x; i < NN / 4; i += 256) d4[i] = s4[i];
    }
    __syncthreads();

    const int lane = threadIdx.x & 31;
    const int wv   = threadIdx.x >> 5;         // 0..7 -> j chunk
    const int m0   = blockIdx.x << 4;          // one 16-row tile per block
    const int l16  = lane & 15;
    const bool hi  = lane >= 16;
    const int jo   = hi ? 16 : 0;              // A-fragment K sub-offset
    const int kofB = hi ? 16 : 0;              // B-fragment K sub-offset
    const int row  = m0 + l16;
    const int jstart = wv << 10;
    const int jend   = jstart + 1024;

    const float gmax = *gmaxp;
    const float fsr  = f_src[row];
    const float sm   = fsr + gmax;
    const float shift = sm > 0.f ? sm : ALPHA * sm;   // >= every score in this row

    const v8f zero8 = {0.f,0.f,0.f,0.f,0.f,0.f,0.f,0.f};
    v8f C[8];
#pragma unroll
    for (int t = 0; t < 8; ++t) C[t] = zero8;
    float psum = 0.f;

    const int* adjr = adj + (size_t)row * NN;

    // software pipeline: preload first adj tile
    int4 q0 = *(const int4*)(adjr + jstart + jo);
    int4 q1 = *(const int4*)(adjr + jstart + jo + 4);
    int4 q2 = *(const int4*)(adjr + jstart + jo + 8);
    int4 q3 = *(const int4*)(adjr + jstart + jo + 12);

#pragma unroll 1
    for (int j0 = jstart; j0 < jend; j0 += 32) {
        const int jb = j0 + jo;
        // prefetch deep within this wave's chunk (wrap keeps address valid)
        __builtin_prefetch(adjr + jstart + ((j0 - jstart + 512) & 1023) + jo, 0, 0);

        // --- region 1: B fragments (L2-resident) + f_dst LDS reads ---------
        v16bf Bf[8];
#pragma unroll
        for (int t = 0; t < 8; ++t)
            Bf[t] = *(const v16bf*)(hbT + (size_t)(t * 16 + l16) * NN + j0 + kofB);
        float4 g0 = *(const float4*)(smem + jb);
        float4 g1 = *(const float4*)(smem + jb + 4);
        float4 g2 = *(const float4*)(smem + jb + 8);
        float4 g3 = *(const float4*)(smem + jb + 12);
        SCHED_FENCE();

        // --- region 2: next-iteration adj loads (HBM), issued after B so
        //     in-order LOADcnt lets us wait on B while adj stays in flight ---
        const int jn = (j0 + 32 < jend) ? (j0 + 32) : jstart;
        int4 n0 = *(const int4*)(adjr + jn + jo);
        int4 n1 = *(const int4*)(adjr + jn + jo + 4);
        int4 n2 = *(const int4*)(adjr + jn + jo + 8);
        int4 n3 = *(const int4*)(adjr + jn + jo + 12);
        SCHED_FENCE();

        // --- region 3: mask + leaky_relu + exp + bf16 pack (VALU only) -----
        int   qa[16] = {q0.x,q0.y,q0.z,q0.w, q1.x,q1.y,q1.z,q1.w,
                        q2.x,q2.y,q2.z,q2.w, q3.x,q3.y,q3.z,q3.w};
        float ga[16] = {g0.x,g0.y,g0.z,g0.w, g1.x,g1.y,g1.z,g1.w,
                        g2.x,g2.y,g2.z,g2.w, g3.x,g3.y,g3.z,g3.w};
        v16bf A;
#pragma unroll
        for (int i = 0; i < 16; ++i) {
            float s = fsr + ga[i];
            float e = s > 0.f ? s : ALPHA * s;             // leaky_relu
            float p = qa[i] > 0 ? __expf(e - shift) : 0.f; // mask + exp (<=1)
            psum += p;
            A[i] = (__bf16)p;
        }
        SCHED_FENCE();

        // --- region 4: WMMA chain ------------------------------------------
#pragma unroll
        for (int t = 0; t < 8; ++t)
            C[t] = __builtin_amdgcn_wmma_f32_16x16x32_bf16(
                false, A, false, Bf[t], (short)0, C[t], false, false);
        SCHED_FENCE();

        q0 = n0; q1 = n1; q2 = n2; q3 = n3;
    }

    // per-wave row denominators: lanes l and l+16 hold complementary halves
    float total = psum + __shfl_xor(psum, 16, 32);

    __syncthreads();                           // all waves done reading f_dst
    if (!hi) psums[wv * 16 + l16] = total;
#pragma unroll
    for (int t = 0; t < 8; ++t)
#pragma unroll
        for (int v = 0; v < 8; ++v)
            smem[wv * 2048 + (v + (hi ? 8 : 0)) * 128 + t * 16 + l16] = C[t][v];
    __syncthreads();

    // cross-wave reduction: 2048 outputs, 8 contiguous per thread
    const int e0 = threadIdx.x * 8;            // element within 16x128 tile
    const int r  = e0 >> 7;                    // row within tile
    float rs = 0.f;
#pragma unroll
    for (int w = 0; w < 8; ++w) rs += psums[w * 16 + r];
    const float inv = rs > 0.f ? 1.f / rs : 0.f;

    float acc[8];
#pragma unroll
    for (int i = 0; i < 8; ++i) acc[i] = 0.f;
#pragma unroll
    for (int w = 0; w < 8; ++w) {
#pragma unroll
        for (int i = 0; i < 8; ++i) acc[i] += smem[w * 2048 + e0 + i];
    }
    float* op = out + (size_t)(m0 + r) * OUTF + (e0 & 127);
#pragma unroll
    for (int i = 0; i < 8; ++i) {
        float o = acc[i] * inv;
        op[i] = o > 0.f ? o : 0.f;             // relu
    }
}

// ---------------------------------------------------------------------------
extern "C" void kernel_launch(void* const* d_in, const int* in_sizes, int n_in,
                              void* d_out, int out_size, void* d_ws, size_t ws_size,
                              hipStream_t stream) {
    (void)in_sizes; (void)n_in; (void)out_size; (void)ws_size;
    const float* X   = (const float*)d_in[0];   // [8192,256] f32
    const int*   adj = (const int*)d_in[1];     // [8192,8192] i32
    const float* W   = (const float*)d_in[2];   // [256,128] f32
    const float* av  = (const float*)d_in[3];   // [256,1] f32
    float* out = (float*)d_out;                 // [8192,128] f32

    char* ws = (char*)d_ws;
    __bf16* WT    = (__bf16*)(ws + OFF_WT);
    __bf16* hbT   = (__bf16*)(ws + OFF_HBT);
    float*  f_src = (float*)(ws + OFF_FSRC);
    float*  f_dst = (float*)(ws + OFF_FDST);
    float*  gmax  = (float*)(ws + OFF_GMAX);

    gat_prep_wt<<<(INF_ * OUTF) / 256, 256, 0, stream>>>(W, WT);
    gat_gemm_h <<<128, 128, 0, stream>>>(X, WT, av, hbT, f_src, f_dst);
    gat_max_fdst<<<1, 1024, 0, stream>>>(f_dst, gmax);
    gat_attn   <<<512, 256, 0, stream>>>(adj, hbT, f_src, f_dst, gmax, out);
}
